// AutoregressiveLSTM_59304908423338
// MI455X (gfx1250) — matrix operand order repaired and checked by reading, not compiled, not run on hardware
//
#include <hip/hip_runtime.h>
#include <hip/hip_bf16.h>

typedef __attribute__((ext_vector_type(16))) __bf16 v16bf;
typedef __attribute__((ext_vector_type(8)))  float  v8f;
typedef int v4i __attribute__((vector_size(4 * sizeof(int))));
typedef __attribute__((address_space(1))) v4i* gv4i_p;   // global b128 chunk
typedef __attribute__((address_space(3))) v4i* lv4i_p;   // LDS b128 chunk

#define B_   2048
#define T_   128
#define V_   36
#define E_   128
#define H_   256
#define G_   1024   // 4*H
#define KTOT 384    // E + H
#define IGNORE_IDX 35

// Fragment-major layouts (wave32 WMMA 16x16x32 bf16):
//  A 16x32: lane = (M&15) + 16*half, element j -> K = half*8 + j + (j&8)
//    inverse: half = (K>>3)&1 ; j = (K&7) | ((K&16)>>1)
//  B 32x16: lane = (N&15) + 16*half, element j -> K = half*8 + j + (j&8)
//    inverse: half = (K>>3)&1 ; j = (K&7) | ((K&16)>>1)
// Each lane's 16 bf16 stored contiguously (32B) -> b128 load pairs.

// ---------------- embedding gather + fragment-major bf16 prepack ----------------
__global__ void k_embed(const int* __restrict__ x, const float* __restrict__ tab,
                        float* __restrict__ emb, __bf16* __restrict__ Aemb) {
  long idx = (long)blockIdx.x * blockDim.x + threadIdx.x;
  const long total = (long)B_ * T_ * E_;
  if (idx >= total) return;
  int  e  = (int)(idx & (E_ - 1));
  long bt = idx >> 7;            // / E_
  int  t  = (int)(bt & (T_ - 1));
  int  b  = (int)(bt >> 7);      // / T_
  int  tok = x[bt];
  float val = tab[tok * E_ + e];
  emb[idx] = val;
  if (t < T_ - 1) {
    int btile = b >> 7;
    int strip = (b >> 4) & 7;
    int Mrow  = b & 15;
    int kc    = e >> 5;
    int kl    = e & 31;
    int half  = (kl >> 3) & 1;
    int jj    = (kl & 7) | ((kl & 16) >> 1);
    int lane  = Mrow + 16 * half;
    long addr = ((((long)t * 16 + btile) * 4 + kc) * 4096) + (strip * 32 + lane) * 16 + jj;
    Aemb[addr] = (__bf16)val;
  }
}

// ------------- pack combined weight fragment-major bf16 + combined bias -------------
__global__ void k_prep(const float* __restrict__ w_ih, const float* __restrict__ w_hh,
                       const float* __restrict__ b_ih, const float* __restrict__ b_hh,
                       __bf16* __restrict__ Wfrag, float* __restrict__ biasc) {
  int idx = blockIdx.x * blockDim.x + threadIdx.x;
  if (idx < KTOT * G_) {
    int k = idx >> 10;          // K index 0..383
    int n = idx & (G_ - 1);     // gate col 0..1023
    float v = (k < E_) ? w_ih[n * E_ + k] : w_hh[n * H_ + (k - E_)];
    int kc   = k >> 5;
    int kk   = k & 31;
    int half = (kk >> 3) & 1;
    int jj   = (kk & 7) | ((kk & 16) >> 1);
    int nfg  = n >> 4;          // global n-fragment 0..63
    int Ncol = n & 15;
    int lane = Ncol + 16 * half;
    Wfrag[((long)(nfg * 12 + kc) * 32 + lane) * 16 + jj] = (__bf16)v;
  }
  if (idx < G_) biasc[idx] = b_ih[idx] + b_hh[idx];
}

__global__ void k_zero(float* __restrict__ p, long n) {
  long i = (long)blockIdx.x * blockDim.x + threadIdx.x;
  if (i < n) p[i] = 0.f;
}

// ---------------- per-step gates GEMM ----------------
// block = 256 threads (8 wave32); tile 128(M=batch) x 128(N=gate cols); K = 384 in 12 chunks.
// Weight fragments for this ntile (contiguous 96KB) are async-copied to LDS once and
// shared by all 8 waves; A fragments are direct global b128 loads (one consumer each).
__global__ __launch_bounds__(256) void k_gates(const __bf16* __restrict__ Aemb,
                                               const __bf16* __restrict__ Ahf,
                                               const __bf16* __restrict__ Wfrag,
                                               float* __restrict__ gates, int t) {
  __shared__ __bf16 Bsh[8 * 12 * 32 * 16];   // 96 KB, same fragment-major order as Wfrag slice

  const int tid   = threadIdx.x;
  const int wave  = tid >> 5;          // strip 0..7 (rows wave*16 .. +15)
  const int lane  = tid & 31;
  const int half  = lane >> 4;
  const int btile = blockIdx.y;        // 0..15
  const int ntile = blockIdx.x;        // 0..7

  {
    const __bf16* wSrc = Wfrag + ((long)ntile * 8 * 12) * 512;   // 49152 bf16 = 96 KB
#if __has_builtin(__builtin_amdgcn_global_load_async_to_lds_b128)
    for (int i = tid; i < 6144; i += 256) {   // 6144 x 16B chunks, uniform trip count
      __builtin_amdgcn_global_load_async_to_lds_b128(
          (gv4i_p)(wSrc + i * 8),
          (lv4i_p)(&Bsh[i * 8]),
          0, 0);
    }
#if __has_builtin(__builtin_amdgcn_s_wait_asynccnt)
    __builtin_amdgcn_s_wait_asynccnt(0);
#else
    asm volatile("s_wait_asynccnt 0" ::: "memory");
#endif
#else
    for (int i = tid; i < 3072; i += 256)
      ((v16bf*)Bsh)[i] = ((const v16bf*)wSrc)[i];
#endif
  }
  __syncthreads();

  const __bf16* aEmbBase = Aemb + (((long)t * 16 + btile) * 4) * 4096 + wave * 512 + lane * 16;
  const __bf16* aHBase   = Ahf  + ((long)btile * 8) * 4096            + wave * 512 + lane * 16;
  const __bf16* bShBase  = Bsh + lane * 16;

  v8f acc[8] = {};

#pragma unroll
  for (int kc = 0; kc < 12; ++kc) {
    const __bf16* ap = (kc < 4) ? (aEmbBase + kc * 4096) : (aHBase + (kc - 4) * 4096);
    v16bf afrag = *(const v16bf*)ap;
    if (kc < 11) {  // prefetch next A fragment
      const __bf16* apn = (kc < 3) ? (aEmbBase + (kc + 1) * 4096) : (aHBase + (kc - 3) * 4096);
      __builtin_prefetch(apn, 0, 1);
    }
#pragma unroll
    for (int nf = 0; nf < 8; ++nf) {
      v16bf bfrag = *(const v16bf*)(bShBase + (nf * 12 + kc) * 512);
      acc[nf] = __builtin_amdgcn_wmma_f32_16x16x32_bf16(
          /*neg_a=*/false, afrag, /*neg_b=*/false, bfrag,
          /*c_mod=*/(short)0, acc[nf], /*reuse_a=*/false, /*reuse_b=*/false);
    }
  }

  // D layout: VGPR r -> M = r + 8*half ; N = lane&15
#pragma unroll
  for (int nf = 0; nf < 8; ++nf) {
    int col = ntile * 128 + nf * 16 + (lane & 15);
#pragma unroll
    for (int r = 0; r < 8; ++r) {
      int row = btile * 128 + wave * 16 + half * 8 + r;
      gates[(long)row * G_ + col] = acc[nf][r];
    }
  }
}

// ---------------- LSTM cell update; writes f32 h and fragment-major bf16 h ----------------
__global__ void k_cell(const float* __restrict__ gates, const float* __restrict__ biasc,
                       float* __restrict__ c, float* __restrict__ h,
                       __bf16* __restrict__ Ahf) {
  int idx = blockIdx.x * blockDim.x + threadIdx.x;   // idx = b*H + j
  int b = idx >> 8;
  int j = idx & (H_ - 1);
  long base = (long)b * G_;
  float ig = gates[base +       j] + biasc[      j];
  float fg = gates[base + 256 + j] + biasc[256 + j];
  float gg = gates[base + 512 + j] + biasc[512 + j];
  float og = gates[base + 768 + j] + biasc[768 + j];
  float iv = 1.f / (1.f + __expf(-ig));
  float fv = 1.f / (1.f + __expf(-fg));
  float gv = tanhf(gg);
  float ov = 1.f / (1.f + __expf(-og));
  float cn = fv * c[idx] + iv * gv;
  float hn = ov * tanhf(cn);
  c[idx] = cn;
  h[idx] = hn;
  // fragment-major bf16 copy for next step's GEMM
  int btile = b >> 7;
  int strip = (b >> 4) & 7;
  int Mrow  = b & 15;
  int kc    = j >> 5;
  int kl    = j & 31;
  int hf    = (kl >> 3) & 1;
  int jj    = (kl & 7) | ((kl & 16) >> 1);
  int lane  = Mrow + 16 * hf;
  Ahf[((long)(btile * 8 + kc) * 4096) + (strip * 32 + lane) * 16 + jj] = (__bf16)hn;
}

// ---------------- fused FC + relu + log_softmax + NLL accumulation ----------------
__global__ __launch_bounds__(256) void k_fc(const float* __restrict__ h,
                                            const float* __restrict__ w_fc,
                                            const float* __restrict__ b_fc,
                                            const int* __restrict__ x,
                                            float* __restrict__ logp_out,
                                            float* __restrict__ nll, int t) {
  __shared__ float wl[V_ * 257];   // pad stride to dodge bank conflicts
  __shared__ float bl[V_];
  int tid = threadIdx.x;
  for (int i = tid; i < V_ * H_; i += 256) {
    int v = i >> 8;
    int k = i & (H_ - 1);
    wl[v * 257 + k] = w_fc[i];
  }
  if (tid < V_) bl[tid] = b_fc[tid];
  __syncthreads();

  int b = blockIdx.x * 256 + tid;
  float lg[V_];
#pragma unroll
  for (int v = 0; v < V_; ++v) lg[v] = bl[v];
  const float* hb = h + (long)b * H_;
  for (int k = 0; k < H_; ++k) {
    float hv = hb[k];
#pragma unroll
    for (int v = 0; v < V_; ++v) lg[v] = fmaf(hv, wl[v * 257 + k], lg[v]);
  }
  float m = 0.f;   // after relu all logits >= 0
#pragma unroll
  for (int v = 0; v < V_; ++v) { lg[v] = fmaxf(lg[v], 0.f); m = fmaxf(m, lg[v]); }
  float s = 0.f;
#pragma unroll
  for (int v = 0; v < V_; ++v) s += __expf(lg[v] - m);
  float lse = m + __logf(s);

  int tgt = x[(long)b * T_ + t + 1];
  float* out = logp_out + ((long)b * (T_ - 1) + t) * V_;
  float picked = 0.f;
#pragma unroll
  for (int v = 0; v < V_; ++v) {
    float lp = lg[v] - lse;
    out[v] = lp;
    if (v == tgt) picked = -lp;
  }
  if (tgt != IGNORE_IDX) nll[b] += picked;
}

// ---------------- per-sample loss ----------------
__global__ void k_sample(const int* __restrict__ x, const float* __restrict__ nll,
                         float* __restrict__ out_sample) {
  int b = blockIdx.x * blockDim.x + threadIdx.x;
  int cnt = 0;
  for (int tt = 0; tt < T_; ++tt) cnt += (x[(long)b * T_ + tt] != IGNORE_IDX);
  out_sample[b] = nll[b] / (float)cnt;
}

// ---------------- mean reduction ----------------
__global__ void k_mean(const float* __restrict__ sl, float* __restrict__ mean_out) {
  __shared__ float red[256];
  int tid = threadIdx.x;
  float s = 0.f;
  for (int i = tid; i < B_; i += 256) s += sl[i];
  red[tid] = s;
  __syncthreads();
  for (int off = 128; off > 0; off >>= 1) {
    if (tid < off) red[tid] += red[tid + off];
    __syncthreads();
  }
  if (tid == 0) mean_out[0] = red[0] / (float)B_;
}

extern "C" void kernel_launch(void* const* d_in, const int* in_sizes, int n_in,
                              void* d_out, int out_size, void* d_ws, size_t ws_size,
                              hipStream_t stream) {
  const int*   x         = (const int*)  d_in[0];
  const float* emb_table = (const float*)d_in[1];
  const float* w_ih      = (const float*)d_in[2];
  const float* w_hh      = (const float*)d_in[3];
  const float* b_ih      = (const float*)d_in[4];
  const float* b_fc      = (const float*)d_in[5];
  const float* b_hh      = (const float*)d_in[6];
  const float* w_fc      = (const float*)d_in[7];

  // d_out layout (return order): log_probs, emb, sample_loss, mean
  float* out  = (float*)d_out;
  float* logp = out;                                  // B*(T-1)*V
  float* emb  = logp + (long)B_ * (T_ - 1) * V_;      // B*T*E
  float* slos = emb  + (long)B_ * T_ * E_;            // B
  float* mean = slos + B_;                            // 1

  // workspace carve (~80 MB); every region 4KB-aligned
  float*  gates = (float*)d_ws;                        // B*G f32            (8 MB)
  float*  c     = gates + (long)B_ * G_;               // B*H f32            (2 MB)
  float*  h     = c     + (long)B_ * H_;               // B*H f32            (2 MB)
  float*  nll   = h     + (long)B_ * H_;               // B f32              (8 KB)
  __bf16* Ahf   = (__bf16*)(nll + B_);                 // B*H bf16 frag      (1 MB)
  float*  biasc = (float*)(Ahf + (long)B_ * H_);       // G f32              (4 KB)
  __bf16* Wfrag = (__bf16*)(biasc + G_);               // KTOT*G bf16 frag   (768 KB)
  __bf16* Aemb  = Wfrag + (long)KTOT * G_;             // (T-1)*B*E bf16 frag(66.6 MB)

  {
    long total = (long)B_ * T_ * E_;
    k_embed<<<(unsigned)((total + 255) / 256), 256, 0, stream>>>(x, emb_table, emb, Aemb);
  }
  k_prep<<<(KTOT * G_ + 255) / 256, 256, 0, stream>>>(w_ih, w_hh, b_ih, b_hh, Wfrag, biasc);
  {
    // zero c, h, nll, Ahf (contiguous; Ahf is B*H bf16 = B*H/2 floats)
    long total = (long)B_ * H_ * 2 + B_ + (long)B_ * H_ / 2;
    k_zero<<<(unsigned)((total + 255) / 256), 256, 0, stream>>>(c, total);
  }

  for (int t = 0; t < T_ - 1; ++t) {
    k_gates<<<dim3(G_ / 128, B_ / 128), 256, 0, stream>>>(Aemb, Ahf, Wfrag, gates, t);
    k_cell<<<(B_ * H_) / 256, 256, 0, stream>>>(gates, biasc, c, h, Ahf);
    k_fc<<<B_ / 256, 256, 0, stream>>>(h, w_fc, b_fc, x, logp, nll, t);
  }

  k_sample<<<B_ / 256, 256, 0, stream>>>(x, nll, slos);
  k_mean<<<1, 256, 0, stream>>>(slos, mean);
}
